// Teacher_model_7464653160858
// MI455X (gfx1250) — compile-verified, matrix operand order checked by reading, not run on hardware
//
#include <hip/hip_runtime.h>
#include <cstdint>
#include <cstddef>

typedef __attribute__((ext_vector_type(16))) _Float16 v16h;
typedef __attribute__((ext_vector_type(8)))  float    v8f;

#define UNITS 1024
#define EMBED 256
#define VOCAB 512
#define BATCH 64
#define TLEN  512
#define G3    3072   // 3*UNITS

// ---------------------------------------------------------------------------
// WMMA helpers (gfx1250: V_WMMA_F32_16X16X32_F16, wave32)
// ---------------------------------------------------------------------------
__device__ __forceinline__ v8f wmma16(v16h a, v16h b, v8f c) {
  // (neg_a, A, neg_b, B, c_mod, C, reuse_a, reuse_b)
  return __builtin_amdgcn_wmma_f32_16x16x32_f16(false, a, false, b, (short)0, c,
                                                false, false);
}

// K index held by VGPR j for a given lane, per the ISA 16-bit A-matrix layout:
//  VGPR0: K=0,1 (lanes 0-15) / K=8,9 (lanes 16-31); VGPR4..7 add +16.
__device__ __forceinline__ int kk_of(int j, int lane) {
  return ((j < 4) ? (2 * j) : (16 + 2 * (j - 4))) + ((lane & 16) ? 8 : 0);
}

// Fragment load from a K-contiguous (row-major [rows][K]) f16 matrix.
// Used for BOTH operands: A rows are M-rows, and weights are pre-packed to
// [N][K] so the B operand has the same per-lane K-contiguous layout
// (lane supplies M or N = rowBase + lane%16). Lowers to 2x global_load_b128.
__device__ __forceinline__ v16h load_frag16(const _Float16* __restrict__ P,
                                            int rowBase, int ld, int k0,
                                            int lane) {
  v16h v;
  const _Float16* base = P + (size_t)(rowBase + (lane & 15)) * ld + k0;
#pragma unroll
  for (int j = 0; j < 8; ++j) {
    int kk = kk_of(j, lane);
    v[2 * j]     = base[kk];
    v[2 * j + 1] = base[kk + 1];
  }
  return v;
}

__device__ __forceinline__ float sigmoidf_fast(float x) {
  return 1.0f / (1.0f + __expf(-x));
}

// ---------------------------------------------------------------------------
// Weight pack: f32 [K][N] row-major  ->  f16 [N][K] (N-major, K-contiguous)
// Tiled 32x32 transpose through LDS; coalesced on both sides.
// ---------------------------------------------------------------------------
__global__ __launch_bounds__(256) void pack_transpose_kernel(
    const float* __restrict__ src, _Float16* __restrict__ dst, int K, int N) {
  __shared__ float tile[32][33];
  const int n0 = blockIdx.x * 32;
  const int k0 = blockIdx.y * 32;
  const int tx = threadIdx.x & 31;   // fast axis
  const int ty = threadIdx.x >> 5;   // 0..7
#pragma unroll
  for (int i = 0; i < 32; i += 8)
    tile[ty + i][tx] = src[(size_t)(k0 + ty + i) * N + (n0 + tx)];
  __syncthreads();
#pragma unroll
  for (int i = 0; i < 32; i += 8)
    dst[(size_t)(n0 + ty + i) * K + (k0 + tx)] = (_Float16)tile[tx][ty + i];
}

// ---------------------------------------------------------------------------
// Elementwise kernels
// ---------------------------------------------------------------------------
__global__ __launch_bounds__(256) void embed_bn_kernel(
    const int* __restrict__ tokens, const float* __restrict__ emb,
    const float* __restrict__ gamma, const float* __restrict__ beta,
    const float* __restrict__ mean, const float* __restrict__ var,
    _Float16* __restrict__ xout) {
  int row = blockIdx.x;        // b*T + t  in [0, 32768)
  int e   = threadIdx.x;       // [0, 256)
  int tok = tokens[row];
  float x = emb[tok * EMBED + e];
  float y = (x - mean[e]) * rsqrtf(var[e] + 1e-3f) * gamma[e] + beta[e];
  xout[(size_t)row * EMBED + e] = (_Float16)y;
}

__global__ __launch_bounds__(256) void zero_state_kernel(
    float* __restrict__ hf32, _Float16* __restrict__ hA,
    _Float16* __restrict__ hB) {
  int i = blockIdx.x * 256 + threadIdx.x;   // grid covers 64*1024
  hf32[i] = 0.0f;
  hA[i] = (_Float16)0.0f;
  hB[i] = (_Float16)0.0f;
}

// ---------------------------------------------------------------------------
// Input-projection GEMM:  Y[M,3072] = X[M,K] * W + bias, W packed as [3072][K].
// Block tile 64x64, 128 threads = 4 waves (wave w -> rows 16w..16w+15).
// No LDS, no barriers; K-loop unrolled x2 with explicit double buffers so each
// half-iteration's loads overlap the other half's WMMAs with zero reg copies.
// K must be a multiple of 64 (256 or 1024 here).
// ---------------------------------------------------------------------------
__global__ __launch_bounds__(128) void proj_gemm_kernel(
    const _Float16* __restrict__ X, const _Float16* __restrict__ Wt,
    const float* __restrict__ bias, float* __restrict__ Y, int K) {
  const int tid  = threadIdx.x;
  const int wave = tid >> 5;
  const int lane = tid & 31;
  const int row0 = blockIdx.x * 64;
  const int col0 = blockIdx.y * 64;

  v8f zero = {0.f, 0.f, 0.f, 0.f, 0.f, 0.f, 0.f, 0.f};
  v8f acc[4];
#pragma unroll
  for (int nt = 0; nt < 4; ++nt) acc[nt] = zero;

  // Buffer set 0: k = 0
  v16h a0 = load_frag16(X, row0 + 16 * wave, K, 0, lane);
  v16h b0[4];
#pragma unroll
  for (int nt = 0; nt < 4; ++nt)
    b0[nt] = load_frag16(Wt, col0 + nt * 16, K, 0, lane);
  v16h a1, b1[4];

  int k0 = 0;
  for (; k0 + 64 < K; k0 += 64) {
    // Load set 1 (k0+32) while consuming set 0 (k0)
    a1 = load_frag16(X, row0 + 16 * wave, K, k0 + 32, lane);
#pragma unroll
    for (int nt = 0; nt < 4; ++nt)
      b1[nt] = load_frag16(Wt, col0 + nt * 16, K, k0 + 32, lane);
#pragma unroll
    for (int nt = 0; nt < 4; ++nt)
      acc[nt] = wmma16(a0, b0[nt], acc[nt]);
    // Load set 0 (k0+64) while consuming set 1 (k0+32)
    a0 = load_frag16(X, row0 + 16 * wave, K, k0 + 64, lane);
#pragma unroll
    for (int nt = 0; nt < 4; ++nt)
      b0[nt] = load_frag16(Wt, col0 + nt * 16, K, k0 + 64, lane);
#pragma unroll
    for (int nt = 0; nt < 4; ++nt)
      acc[nt] = wmma16(a1, b1[nt], acc[nt]);
  }
  // Tail: pair (k0, k0+32)
  a1 = load_frag16(X, row0 + 16 * wave, K, k0 + 32, lane);
#pragma unroll
  for (int nt = 0; nt < 4; ++nt)
    b1[nt] = load_frag16(Wt, col0 + nt * 16, K, k0 + 32, lane);
#pragma unroll
  for (int nt = 0; nt < 4; ++nt)
    acc[nt] = wmma16(a0, b0[nt], acc[nt]);
#pragma unroll
  for (int nt = 0; nt < 4; ++nt)
    acc[nt] = wmma16(a1, b1[nt], acc[nt]);

  // C/D layout: N = lane%16; M = r (+8 for lanes 16-31).
  const int rbase = (lane & 16) ? 8 : 0;
  const int nn = lane & 15;
#pragma unroll
  for (int nt = 0; nt < 4; ++nt) {
    int col = col0 + nt * 16 + nn;
    float bv = bias[col];
#pragma unroll
    for (int r = 0; r < 8; ++r) {
      int row = row0 + 16 * wave + r + rbase;
      Y[(size_t)row * G3 + col] = acc[nt][r] + bv;
    }
  }
}

// ---------------------------------------------------------------------------
// One GRU timestep:  inner = h_in * Wr (+b_rec); gates; h update; emit seq.
// Wr packed as [3072][1024] (N-major). Grid 64 blocks, each owning 16 columns
// of every gate; 128 threads = 4 waves (wave w -> batch rows 16w..16w+15),
// 3 accumulators per wave. Pure global->WMMA loop, x2 unrolled double buffer.
// h_f16 is ping-ponged across steps (the GEMM reads ALL columns of h_in).
// ---------------------------------------------------------------------------
__global__ __launch_bounds__(128) void gru_step_kernel(
    const _Float16* __restrict__ h16_in, _Float16* __restrict__ h16_out,
    float* __restrict__ h32, const float* __restrict__ xproj,
    const _Float16* __restrict__ Wrt, const float* __restrict__ brec,
    _Float16* __restrict__ seq_out, int t) {
  const int tid  = threadIdx.x;
  const int wave = tid >> 5;
  const int lane = tid & 31;
  const int u0 = blockIdx.x * 16;      // column base within each gate

  v8f zero = {0.f, 0.f, 0.f, 0.f, 0.f, 0.f, 0.f, 0.f};
  v8f acc[3];
#pragma unroll
  for (int g = 0; g < 3; ++g) acc[g] = zero;

  // Buffer set 0: k = 0
  v16h a0 = load_frag16(h16_in, 16 * wave, UNITS, 0, lane);
  v16h b0[3];
#pragma unroll
  for (int g = 0; g < 3; ++g)
    b0[g] = load_frag16(Wrt, g * UNITS + u0, UNITS, 0, lane);
  v16h a1, b1[3];

  int k0 = 0;
  for (; k0 + 64 < UNITS; k0 += 64) {
    a1 = load_frag16(h16_in, 16 * wave, UNITS, k0 + 32, lane);
#pragma unroll
    for (int g = 0; g < 3; ++g)
      b1[g] = load_frag16(Wrt, g * UNITS + u0, UNITS, k0 + 32, lane);
#pragma unroll
    for (int g = 0; g < 3; ++g)
      acc[g] = wmma16(a0, b0[g], acc[g]);
    a0 = load_frag16(h16_in, 16 * wave, UNITS, k0 + 64, lane);
#pragma unroll
    for (int g = 0; g < 3; ++g)
      b0[g] = load_frag16(Wrt, g * UNITS + u0, UNITS, k0 + 64, lane);
#pragma unroll
    for (int g = 0; g < 3; ++g)
      acc[g] = wmma16(a1, b1[g], acc[g]);
  }
  a1 = load_frag16(h16_in, 16 * wave, UNITS, k0 + 32, lane);
#pragma unroll
  for (int g = 0; g < 3; ++g)
    b1[g] = load_frag16(Wrt, g * UNITS + u0, UNITS, k0 + 32, lane);
#pragma unroll
  for (int g = 0; g < 3; ++g)
    acc[g] = wmma16(a0, b0[g], acc[g]);
#pragma unroll
  for (int g = 0; g < 3; ++g)
    acc[g] = wmma16(a1, b1[g], acc[g]);

  // Gates + state update. This block exclusively owns h columns [u0, u0+16),
  // so h32 may be updated in place; h16 goes to the ping-pong 'out' buffer.
  const int rbase = (lane & 16) ? 8 : 0;
  const int nn = lane & 15;
  const int u = u0 + nn;
  const float bz = brec[u];
  const float br = brec[UNITS + u];
  const float bh = brec[2 * UNITS + u];
#pragma unroll
  for (int r = 0; r < 8; ++r) {
    int row = 16 * wave + r + rbase;                    // batch index 0..63
    const float* xp = xproj + ((size_t)row * TLEN + t) * G3;
    float z  = sigmoidf_fast(xp[u] + acc[0][r] + bz);
    float rg = sigmoidf_fast(xp[UNITS + u] + acc[1][r] + br);
    float hh = tanhf(xp[2 * UNITS + u] + rg * (acc[2][r] + bh));
    float hprev = h32[row * UNITS + u];
    float hnew  = z * hprev + (1.0f - z) * hh;
    h32[row * UNITS + u]     = hnew;
    h16_out[row * UNITS + u] = (_Float16)hnew;
    seq_out[((size_t)row * TLEN + t) * UNITS + u] = (_Float16)hnew;
  }
}

// ---------------------------------------------------------------------------
// Dense + softmax:  s[64,1024] -> [64,4,256] . W[256,512] + b -> softmax(512)
// One block per (b,f) row; thread j covers columns j and j+256.
// ---------------------------------------------------------------------------
__global__ __launch_bounds__(256) void dense_softmax_kernel(
    const float* __restrict__ h32, const float* __restrict__ Wd,
    const float* __restrict__ bd, float* __restrict__ out) {
  __shared__ float sX[256];
  __shared__ float red[256];
  const int blk = blockIdx.x;       // b*4 + f
  const int tid = threadIdx.x;
  const int b = blk >> 2, f = blk & 3;

  sX[tid] = h32[b * UNITS + f * 256 + tid];
  __syncthreads();

  float a0 = bd[tid], a1 = bd[tid + 256];
  for (int d = 0; d < 256; ++d) {
    float xv = sX[d];
    a0 += xv * Wd[d * VOCAB + tid];
    a1 += xv * Wd[d * VOCAB + tid + 256];
  }
  // max-reduce
  red[tid] = fmaxf(a0, a1);
  __syncthreads();
  for (int s = 128; s > 0; s >>= 1) {
    if (tid < s) red[tid] = fmaxf(red[tid], red[tid + s]);
    __syncthreads();
  }
  float mx = red[0];
  __syncthreads();
  float e0 = __expf(a0 - mx), e1 = __expf(a1 - mx);
  red[tid] = e0 + e1;
  __syncthreads();
  for (int s = 128; s > 0; s >>= 1) {
    if (tid < s) red[tid] += red[tid + s];
    __syncthreads();
  }
  float inv = 1.0f / red[0];
  out[(size_t)blk * VOCAB + tid]       = e0 * inv;
  out[(size_t)blk * VOCAB + tid + 256] = e1 * inv;
}

// ---------------------------------------------------------------------------
// Host launcher
// ---------------------------------------------------------------------------
static inline void* carve(char*& p, size_t bytes) {
  void* r = (void*)p;
  p += (bytes + 255) & ~(size_t)255;
  return r;
}

extern "C" void kernel_launch(void* const* d_in, const int* in_sizes, int n_in,
                              void* d_out, int out_size, void* d_ws, size_t ws_size,
                              hipStream_t stream) {
  (void)in_sizes; (void)n_in; (void)out_size; (void)ws_size;
  const int*   tokens = (const int*)d_in[0];
  const float* emb    = (const float*)d_in[1];
  const float* gamma  = (const float*)d_in[2];
  const float* beta   = (const float*)d_in[3];
  const float* mean   = (const float*)d_in[4];
  const float* var    = (const float*)d_in[5];
  const float* gk[3] = {(const float*)d_in[6], (const float*)d_in[9],  (const float*)d_in[12]};
  const float* gr[3] = {(const float*)d_in[7], (const float*)d_in[10], (const float*)d_in[13]};
  const float* gb[3] = {(const float*)d_in[8], (const float*)d_in[11], (const float*)d_in[14]};
  const float* dense_w = (const float*)d_in[15];
  const float* dense_b = (const float*)d_in[16];
  float* out = (float*)d_out;

  const int M = BATCH * TLEN;           // 32768 rows
  char* p = (char*)d_ws;
  _Float16* xf16 = (_Float16*)carve(p, (size_t)M * EMBED * 2);
  _Float16* wk16[3], *wr16[3];
  wk16[0] = (_Float16*)carve(p, (size_t)G3 * EMBED * 2);   // packed [3072][256]
  wk16[1] = (_Float16*)carve(p, (size_t)G3 * UNITS * 2);   // packed [3072][1024]
  wk16[2] = (_Float16*)carve(p, (size_t)G3 * UNITS * 2);
  for (int l = 0; l < 3; ++l) wr16[l] = (_Float16*)carve(p, (size_t)G3 * UNITS * 2);
  float*    xproj = (float*)carve(p, (size_t)M * G3 * 4);
  _Float16* seqA  = (_Float16*)carve(p, (size_t)M * UNITS * 2);
  _Float16* seqB  = (_Float16*)carve(p, (size_t)M * UNITS * 2);
  float*    h32   = (float*)carve(p, (size_t)BATCH * UNITS * 4);
  _Float16* hA    = (_Float16*)carve(p, (size_t)BATCH * UNITS * 2);
  _Float16* hB    = (_Float16*)carve(p, (size_t)BATCH * UNITS * 2);

  // 1) Pack weights: f32 [K][3072] -> f16 [3072][K]
  {
    dim3 blk(256);
    dim3 g0(G3 / 32, EMBED / 32), g1(G3 / 32, UNITS / 32);
    pack_transpose_kernel<<<g0, blk, 0, stream>>>(gk[0], wk16[0], EMBED, G3);
    pack_transpose_kernel<<<g1, blk, 0, stream>>>(gk[1], wk16[1], UNITS, G3);
    pack_transpose_kernel<<<g1, blk, 0, stream>>>(gk[2], wk16[2], UNITS, G3);
    for (int l = 0; l < 3; ++l)
      pack_transpose_kernel<<<g1, blk, 0, stream>>>(gr[l], wr16[l], UNITS, G3);
  }

  // 2) Embedding + BatchNorm -> x_f16 [M, 256]
  embed_bn_kernel<<<M, 256, 0, stream>>>(tokens, emb, gamma, beta, mean, var, xf16);

  // 3) Zero initial state
  zero_state_kernel<<<(BATCH * UNITS) / 256, 256, 0, stream>>>(h32, hA, hB);

  // 4) Three GRU layers
  const _Float16* layer_in[3] = {xf16, seqA, seqB};
  _Float16* layer_out[3] = {seqA, seqB, seqA};  // layer2 seq unused; clobber seqA
  const int layer_K[3] = {EMBED, UNITS, UNITS};
  _Float16* hcur = hA;
  _Float16* hnxt = hB;
  for (int l = 0; l < 3; ++l) {
    dim3 pg(M / 64, G3 / 64);
    proj_gemm_kernel<<<pg, 128, 0, stream>>>(layer_in[l], wk16[l], gb[l], xproj,
                                             layer_K[l]);
    const float* brec = gb[l] + G3;  // gru*_b row 1
    for (int t = 0; t < TLEN; ++t) {
      gru_step_kernel<<<UNITS / 16, 128, 0, stream>>>(hcur, hnxt, h32, xproj,
                                                      wr16[l], brec, layer_out[l], t);
      _Float16* tmp = hcur; hcur = hnxt; hnxt = tmp;
    }
  }

  // 5) Dense + softmax on final state
  dense_softmax_kernel<<<BATCH * 4, 256, 0, stream>>>(h32, dense_w, dense_b, out);
}
